// QLSTMGen_65481071405340
// MI455X (gfx1250) — matrix operand order repaired and checked by reading, not compiled, not run on hardware
//
#include <hip/hip_runtime.h>
#include <hip/hip_bf16.h>
#include <math.h>

// Problem dims (fixed by the reference)
#define T_STEPS 512
#define B_SZ    128
#define D_SZ    512
#define H_SZ    512
#define NQ_SZ   16
#define G_SZ    4
#define CIN     (D_SZ + H_SZ)   // 1024
#define NBLK    32              // persistent workgroups (4 B-tiles x 8 H-blocks)

typedef __attribute__((ext_vector_type(16))) __bf16 v16bf;
typedef __attribute__((ext_vector_type(8)))  __bf16 v8bf;
typedef __attribute__((ext_vector_type(8)))  float  v8f;

static __device__ __forceinline__ v8f wmma_bf16(v16bf a, v16bf b, v8f c) {
  return __builtin_amdgcn_wmma_f32_16x16x32_bf16(
      false, a, false, b, (short)0, c, false, false);
}

static __device__ __forceinline__ v8f vzero8() {
  v8f z;
#pragma unroll
  for (int i = 0; i < 8; ++i) z[i] = 0.0f;
  return z;
}

static __device__ __forceinline__ v16bf join8(v8bf lo, v8bf hi) {
  v16bf r;
#pragma unroll
  for (int i = 0; i < 8; ++i) { r[i] = lo[i]; r[i + 8] = hi[i]; }
  return r;
}

static __device__ __forceinline__ float sigmoidf_(float x) {
  return 1.0f / (1.0f + __expf(-x));
}

// Device-wide barrier for the persistent kernel: L2 atomics + s_sleep spin.
static __device__ __forceinline__ void gsync(unsigned* bar, unsigned* gen) {
  __syncthreads();
  __threadfence();                       // release this block's writes to agent scope
  if (threadIdx.x == 0) {
    const unsigned target = ++(*gen) * NBLK;
    __hip_atomic_fetch_add(bar, 1u, __ATOMIC_RELEASE, __HIP_MEMORY_SCOPE_AGENT);
    while (__hip_atomic_load(bar, __ATOMIC_ACQUIRE, __HIP_MEMORY_SCOPE_AGENT) < target)
      __builtin_amdgcn_s_sleep(2);
  }
  __syncthreads();
  __threadfence();                       // acquire other blocks' writes
}

// ---------------------------------------------------------------------------
// Prep: weights -> [N][K] bf16 (B-fragment order), zero hx/cx/barrier.
// ---------------------------------------------------------------------------
#define N_WLT  (G_SZ * NQ_SZ * CIN)        // 65536
#define N_W1T  (G_SZ * H_SZ * 32)          // 65536 (K padded 16->32)
#define N_W2T  (G_SZ * H_SZ * H_SZ)        // 1048576
#define N_ST   (B_SZ * H_SZ)               // 65536
#define N_PREP (N_WLT + N_W1T + N_W2T + 2 * N_ST + 1)

__global__ void k_prep(const float* __restrict__ Wl, const float* __restrict__ W1,
                       const float* __restrict__ W2,
                       __bf16* __restrict__ WlT, __bf16* __restrict__ W1T,
                       __bf16* __restrict__ W2T,
                       float* __restrict__ hx, float* __restrict__ cx,
                       unsigned* __restrict__ bar) {
  long idx = (long)blockIdx.x * blockDim.x + threadIdx.x;
  if (idx < N_WLT) {
    int c = (int)(idx & (CIN - 1)); int gq = (int)(idx >> 10);
    int q = gq & (NQ_SZ - 1); int g = gq >> 4;
    WlT[idx] = (__bf16)Wl[(g * CIN + c) * NQ_SZ + q];
  } else if (idx < N_WLT + N_W1T) {
    long i = idx - N_WLT;
    int k = (int)(i & 31); int gh = (int)(i >> 5);
    int h = gh & (H_SZ - 1); int g = gh >> 9;
    W1T[i] = (k < NQ_SZ) ? (__bf16)W1[(g * NQ_SZ + k) * H_SZ + h] : (__bf16)0.0f;
  } else if (idx < N_WLT + N_W1T + N_W2T) {
    long i = idx - (N_WLT + N_W1T);
    int k = (int)(i & (H_SZ - 1)); int gh = (int)(i >> 9);
    int h = gh & (H_SZ - 1); int g = gh >> 9;
    W2T[i] = (__bf16)W2[((long)g * H_SZ + k) * H_SZ + h];
  } else if (idx < N_WLT + N_W1T + N_W2T + N_ST) {
    hx[idx - (N_WLT + N_W1T + N_W2T)] = 0.0f;
  } else if (idx < N_WLT + N_W1T + N_W2T + 2 * N_ST) {
    cx[idx - (N_WLT + N_W1T + N_W2T + N_ST)] = 0.0f;
  } else if (idx == N_WLT + N_W1T + N_W2T + 2 * N_ST) {
    bar[0] = 0u;
  }
}

// ---------------------------------------------------------------------------
// Persistent LSTM kernel: 32 blocks x 256 threads, launched once.
//   per step: [z + h1 fused in-block] -> gsync -> [g4 + cell update] -> gsync
// ---------------------------------------------------------------------------
struct G4Buf {                       // double-buffered WMMA staging tiles
  __bf16 A[2][G_SZ][32][32];         // h1 tiles   [rows][K]  (2 x 8 KB)
  __bf16 B[2][G_SZ][64][32];         // W2T tiles  [N][K]     (2 x 16 KB)
};
struct ZBuf {
  float  zred[8][32][8];             // per-wave partial z accumulators (8 KB)
  __bf16 zt[16][32];                 // z tile as A-fragment source, K padded
};
union __align__(64) SMem {
  G4Buf g;
  ZBuf  z;
};

__global__ __launch_bounds__(256, 1) void k_lstm(
    const float* __restrict__ inputs,
    const __bf16* __restrict__ WlT, const float* __restrict__ bl,
    const __bf16* __restrict__ W1T, const float* __restrict__ b1,
    const __bf16* __restrict__ W2T, const float* __restrict__ b2,
    __bf16* h1, float* hx, float* cx,
    float* __restrict__ out, unsigned* bar) {
  __shared__ SMem sm;

  const int tid  = threadIdx.x;
  const int w    = tid >> 5;
  const int lane = tid & 31;
  const int half = lane >> 4;
  const int l16  = lane & 15;
  const int blk  = blockIdx.x;            // 0..31

  // z/h1 phase mapping: block <-> (gate, 16-row tile of B)
  const int zg  = blk >> 3;
  const int zmt = blk & 7;
  // g4 phase mapping: block <-> 32(B) x 64(H) output tile
  const int Bt = (blk & 3) * 32;
  const int Hs = (blk >> 2) * 64;

  unsigned gen = 0;

  // async stage of one K-slice of the 4-gate W2 GEMM into LDS buffer `buf`
  auto issue_g4 = [&](int buf, int k0) {
    const unsigned aBase = (unsigned)(size_t)&sm.g.A[buf][0][0][0];
    const unsigned bBase = (unsigned)(size_t)&sm.g.B[buf][0][0][0];
#pragma unroll
    for (int i = 0; i < 2; ++i) {        // A: 4g * 32 rows * 64B = 512 * 16B
      int idx = tid + i * 256;
      int gg = idx >> 7; int rem = idx & 127; int r = rem >> 2; int c = rem & 3;
      unsigned long long src = (unsigned long long)(size_t)
          (h1 + ((gg * B_SZ + Bt + r) * H_SZ + k0) + c * 8);
      unsigned dst = aBase + idx * 16;
      asm volatile("global_load_async_to_lds_b128 %0, %1, off"
                   :: "v"(dst), "v"(src) : "memory");
    }
#pragma unroll
    for (int i = 0; i < 4; ++i) {        // B: 4g * 64 N-rows * 64B = 1024 * 16B
      int idx = tid + i * 256;
      int gg = idx >> 8; int rem = idx & 255; int n = rem >> 2; int c = rem & 3;
      unsigned long long src = (unsigned long long)(size_t)
          (W2T + ((gg * H_SZ + Hs + n) * H_SZ + k0) + c * 8);
      unsigned dst = bBase + idx * 16;
      asm volatile("global_load_async_to_lds_b128 %0, %1, off"
                   :: "v"(dst), "v"(src) : "memory");
    }
  };

#pragma unroll 1
  for (int t = 0; t < T_STEPS; ++t) {
    const float* xt = inputs + (long)t * B_SZ * D_SZ;
    float* out_t = out + (long)t * N_ST;

    // ---------------- phase Z: z(g,mt) = concat(x,hx) @ Wl + bl ------------
    {
      v8f zacc = vzero8();
      const int kbase = w * 128;               // 8 waves split K=1024
#pragma unroll
      for (int kk = 0; kk < 4; ++kk) {
        const int k0 = kbase + kk * 32;
        const float* abase = (k0 < D_SZ)
            ? (xt + (zmt * 16 + l16) * D_SZ + k0)
            : (hx + (zmt * 16 + l16) * H_SZ + (k0 - D_SZ));
        v16bf a;
#pragma unroll
        for (int e = 0; e < 8; ++e) a[e]     = (__bf16)abase[16 * half + e];
#pragma unroll
        for (int e = 0; e < 8; ++e) a[e + 8] = (__bf16)abase[16 * half + 8 + e];
        v16bf b = *(const v16bf*)(WlT + ((zg * NQ_SZ + l16) * CIN + k0 + 16 * half));
        zacc = wmma_bf16(a, b, zacc);
      }
      *(v8f*)&sm.z.zred[w][lane][0] = zacc;
      __syncthreads();
      if (w == 0) {                            // cross-wave K reduction
        v8f s = vzero8();
#pragma unroll
        for (int ww = 0; ww < 8; ++ww) {
          v8f p = *(const v8f*)&sm.z.zred[ww][lane][0];
#pragma unroll
          for (int r = 0; r < 8; ++r) s[r] += p[r];
        }
        const float bias = bl[zg * NQ_SZ + l16];
#pragma unroll
        for (int r = 0; r < 8; ++r) {          // C layout: M = 8*half + r
          sm.z.zt[8 * half + r][l16]      = (__bf16)(s[r] + bias);
          sm.z.zt[8 * half + r][16 + l16] = (__bf16)0.0f;   // K padding
        }
      }
      __syncthreads();
    }

    // ---------------- phase H1: h1(g,mt,:) = relu(z @ W1 + b1) -------------
    {
      const __bf16* ap = &sm.z.zt[l16][0];
      v16bf a = join8(*(const v8bf*)(ap + 16 * half),
                      *(const v8bf*)(ap + 16 * half + 8));
#pragma unroll
      for (int i = 0; i < 4; ++i) {            // 32 N-tiles over 8 waves
        const int nt = w * 4 + i;
        const int hcol = nt * 16 + l16;
        v16bf b = *(const v16bf*)(W1T + ((zg * H_SZ + hcol) * 32 + 16 * half));
        v8f acc = vzero8();
        acc = wmma_bf16(a, b, acc);
        const float bias = b1[zg * H_SZ + hcol];
#pragma unroll
        for (int r = 0; r < 8; ++r) {
          float v = acc[r] + bias;
          v = v > 0.0f ? v : 0.0f;
          h1[(zg * B_SZ + (zmt * 16 + 8 * half + r)) * H_SZ + hcol] = (__bf16)v;
        }
      }
    }
    gsync(bar, &gen);                          // h1 visible device-wide

    // ---------------- phase G4: g4 = h1 @ W2 + b2, fused cell update -------
    {
      const int wm = w >> 2, wn = w & 3;       // wave's 16x16 sub-tile
      v8f acc[G_SZ];
#pragma unroll
      for (int g = 0; g < G_SZ; ++g) acc[g] = vzero8();

      issue_g4(0, 0);                          // prime buffer 0
      int kb = 0;
#pragma unroll 1
      for (int k0 = 0; k0 < H_SZ; k0 += 32, kb ^= 1) {
        if (k0 + 32 < H_SZ) {
          issue_g4(kb ^ 1, k0 + 32);           // prefetch next slice async
          asm volatile("s_wait_asynccnt 0x6" ::: "memory");  // cur slice done
        } else {
          asm volatile("s_wait_asynccnt 0x0" ::: "memory");
        }
        __syncthreads();
#pragma unroll
        for (int g = 0; g < G_SZ; ++g) {
          const __bf16* ap = &sm.g.A[kb][g][wm * 16 + l16][0];
          v16bf a = join8(*(const v8bf*)(ap + 16 * half),
                          *(const v8bf*)(ap + 16 * half + 8));
          v16bf b = *(const v16bf*)&sm.g.B[kb][g][wn * 16 + l16][16 * half];
          acc[g] = wmma_bf16(a, b, acc[g]);
        }
        __syncthreads();
      }

      const int hcol = Hs + wn * 16 + l16;
      const float bF = b2[0 * H_SZ + hcol];
      const float bI = b2[1 * H_SZ + hcol];
      const float bG = b2[2 * H_SZ + hcol];
      const float bO = b2[3 * H_SZ + hcol];
#pragma unroll
      for (int r = 0; r < 8; ++r) {
        const int  brow = Bt + wm * 16 + 8 * half + r;
        const long idx  = (long)brow * H_SZ + hcol;
        const float f  = sigmoidf_(acc[0][r] + bF);
        const float i  = sigmoidf_(acc[1][r] + bI);
        const float gg = tanhf(acc[2][r] + bG);
        const float o  = sigmoidf_(acc[3][r] + bO);
        const float c_new = f * cx[idx] + i * gg;
        const float h_new = o * tanhf(c_new);
        cx[idx] = c_new;
        hx[idx] = h_new;
        out_t[idx] = h_new;
      }
    }
    gsync(bar, &gen);                          // hx visible for next step's Z
  }

  // Epilogue: final hx, cx -> d_out tail (after last gsync, state is global)
  float* tail = out + (long)T_STEPS * N_ST;
#pragma unroll 1
  for (int i = tid; i < N_ST / NBLK; i += 256) {
    const int idx = blk * (N_ST / NBLK) + i;
    tail[idx]        = hx[idx];
    tail[N_ST + idx] = cx[idx];
  }
}

extern "C" void kernel_launch(void* const* d_in, const int* in_sizes, int n_in,
                              void* d_out, int out_size, void* d_ws, size_t ws_size,
                              hipStream_t stream) {
  const float* inputs = (const float*)d_in[0];
  const float* Wl = (const float*)d_in[1];
  const float* bl = (const float*)d_in[2];
  const float* W1 = (const float*)d_in[3];
  const float* b1 = (const float*)d_in[4];
  const float* W2 = (const float*)d_in[5];
  const float* b2 = (const float*)d_in[6];

  char* ws = (char*)d_ws;
  size_t off = 0;
  __bf16* WlT = (__bf16*)(ws + off); off += (size_t)N_WLT * 2;          // 128 KB
  __bf16* W1T = (__bf16*)(ws + off); off += (size_t)N_W1T * 2;          // 128 KB
  __bf16* W2T = (__bf16*)(ws + off); off += (size_t)N_W2T * 2;          // 2 MB
  __bf16* h1  = (__bf16*)(ws + off); off += (size_t)G_SZ * B_SZ * H_SZ * 2;
  float*  hx  = (float*)(ws + off);  off += (size_t)N_ST * 4;
  float*  cx  = (float*)(ws + off);  off += (size_t)N_ST * 4;
  unsigned* bar = (unsigned*)(ws + off); off += 64;
  float* out = (float*)d_out;

  k_prep<<<(N_PREP + 255) / 256, 256, 0, stream>>>(Wl, W1, W2, WlT, W1T, W2T,
                                                   hx, cx, bar);
  k_lstm<<<NBLK, 256, 0, stream>>>(inputs, WlT, bl, W1T, b1, W2T, b2,
                                   h1, hx, cx, out, bar);
}
